// BiAttentionMRU_26147760898613
// MI455X (gfx1250) — compile-verified, matrix operand order checked by reading, not run on hardware
//
#include <hip/hip_runtime.h>
#include <hip/hip_bf16.h>

// ---------------------------------------------------------------------------
// BiAttentionMRU forward for MI455X (gfx1250, wave32, WMMA).
// All dense GEMMs run through v_wmma_f32_16x16x32_f16 (f16 in, f32 acc),
// register-blocked NT output tiles per wave. K is a template constant
// (320 or 32) so the K loop fully unrolls: loads hoist across WMMAs and
// waits become partial (no s_wait_loadcnt 0x0 serialization).
// ---------------------------------------------------------------------------

#define DIMV 300
#define KP   320
#define TART 2000
#define BB   16
#define TQ   30
#define TOPT 16

typedef _Float16 v16h __attribute__((ext_vector_type(16)));
typedef _Float16 v8h  __attribute__((ext_vector_type(8)));
typedef float    v8f  __attribute__((ext_vector_type(8)));

// ---------------------------------------------------------------------------
// WMMA GEMM:  C[M,N] = act( A[M,KPC] * W[N,KPC]^T + bias )
// A, W row-major f16, row stride == KPC (zero padded). C f32, row stride ldC.
// Batched via gridDim.z. Block = 4 waves; each wave owns one M-tile
// (mt = blockIdx.x*4 + wave) and NT consecutive N-tiles (group blockIdx.y).
// Per ISA fragment layout, lane holds row (lane&15) and its 16 halves are two
// contiguous 8-half runs at K = k0 + (lane>>4)*8 and +16 -> two b128 loads.
// ---------------------------------------------------------------------------
template<int NT, int KPC>
__global__ __launch_bounds__(128)
void k_gemm_wmma(const _Float16* __restrict__ A, long long aStride,
                 const _Float16* __restrict__ W, long long wStride,
                 const float* __restrict__ bias,
                 float* __restrict__ C, long long cStride,
                 int M, int N, int ldC, int act)
{
    const int lane = threadIdx.x & 31;
    const int wave = threadIdx.x >> 5;
    const int mt   = blockIdx.x * 4 + wave;
    if (mt * 16 >= M) return;                 // wave-uniform exit, no barriers
    const int ntg  = blockIdx.y;              // group of NT n-tiles
    const int b    = blockIdx.z;

    const _Float16* Ab = A + (long long)b * aStride;
    const _Float16* Wb = W + (long long)b * wStride;
    float*          Cb = C + (long long)b * cStride;

    const int arow  = mt * 16 + (lane & 15);
    const int khalf = (lane >> 4) << 3;       // 0 or 8
    const _Float16* ap  = Ab + (long long)arow * KPC + khalf;
    const _Float16* wp0 = Wb + (long long)(ntg * NT * 16 + (lane & 15)) * KPC + khalf;

    const v8f vzero = {0.f, 0.f, 0.f, 0.f, 0.f, 0.f, 0.f, 0.f};
    v8f acc[NT];
#pragma unroll
    for (int j = 0; j < NT; ++j) acc[j] = vzero;

#pragma unroll
    for (int k0 = 0; k0 < KPC; k0 += 32) {
        v8h alo = *(const v8h*)(ap + k0);
        v8h ahi = *(const v8h*)(ap + k0 + 16);
        v8h blo[NT], bhi[NT];
#pragma unroll
        for (int j = 0; j < NT; ++j) {        // distinct regs per fragment
            const _Float16* wp = wp0 + (long long)j * 16 * KPC + k0;
            blo[j] = *(const v8h*)(wp);
            bhi[j] = *(const v8h*)(wp + 16);
        }
        v16h av = __builtin_shufflevector(alo, ahi,
                    0,1,2,3,4,5,6,7,8,9,10,11,12,13,14,15);
#pragma unroll
        for (int j = 0; j < NT; ++j) {
            v16h bv = __builtin_shufflevector(blo[j], bhi[j],
                        0,1,2,3,4,5,6,7,8,9,10,11,12,13,14,15);
            acc[j] = __builtin_amdgcn_wmma_f32_16x16x32_f16(
                         false, av, false, bv, (short)0, acc[j], false, false);
        }
    }

    const int rbase = mt * 16 + ((lane >> 4) << 3);
#pragma unroll
    for (int j = 0; j < NT; ++j) {
        const int col = (ntg * NT + j) * 16 + (lane & 15);
        if (col < N) {
            const float bvv = bias ? bias[col] : 0.0f;
#pragma unroll
            for (int r = 0; r < 8; ++r) {
                float v = acc[j][r] + bvv;
                if (act == 1)      v = v > 0.f ? v : 0.f;   // relu
                else if (act == 2) v = tanhf(v);            // tanh
                Cb[(long long)(rbase + r) * ldC + col] = v;
            }
        }
    }
}

// --- embedding gather -> f32 (300) and padded f16 (320, pad zero) ----------
__global__ __launch_bounds__(320)
void k_gather(const int* __restrict__ ids, const float* __restrict__ emb,
              float* __restrict__ f32out, _Float16* __restrict__ f16out)
{
    const int row = blockIdx.x, c = threadIdx.x;
    const long long id = ids[row];
    float v = 0.f;
    if (c < DIMV) {
        v = emb[id * DIMV + c];
        f32out[(long long)row * DIMV + c] = v;
    }
    f16out[(long long)row * KP + c] = (_Float16)v;
}

// --- weight cast f32 (300x300) -> f16 (320x320, zero padded) ---------------
__global__ __launch_bounds__(320)
void k_cast_w(const float* __restrict__ w, _Float16* __restrict__ out)
{
    const int r = blockIdx.x, c = threadIdx.x;
    const float v = (r < DIMV && c < DIMV) ? w[r * DIMV + c] : 0.f;
    out[r * KP + c] = (_Float16)v;
}

// --- ContractExpand group sum: xs[b,g,d] = sum_{j<r} art[b, gi*r+j, d] -----
__global__ __launch_bounds__(320)
void k_group_sum(const float* __restrict__ art, _Float16* __restrict__ xs,
                 int r, int g)
{
    const int row = blockIdx.x;            // b*g + gi
    const int c   = threadIdx.x;
    const int b = row / g, gi = row % g;
    float s = 0.f;
    if (c < DIMV) {
        const float* p = art + ((long long)(b * TART + gi * r)) * DIMV + c;
        for (int j = 0; j < r; ++j) s += p[(long long)j * DIMV];
    }
    xs[(long long)row * KP + c] = (_Float16)((c < DIMV) ? s : 0.f);
}

// --- gate MLP + sequential MRU scan; emits art_enc f32 + padded f16 --------
// Per-thread sequential t-loop with a dependent recurrence: prefetch the
// z/o streams ~8 steps ahead (global_prefetch_b8) to hide load latency.
__global__ __launch_bounds__(320)
void k_gate_scan(const float* __restrict__ h0, const float* __restrict__ h1,
                 const float* __restrict__ h2, const float* __restrict__ h3,
                 const float* __restrict__ h4,
                 const float* __restrict__ z, const float* __restrict__ o,
                 const float* __restrict__ mr1W, const float* __restrict__ mr1b,
                 const float* __restrict__ mr2W, const float* __restrict__ mr2b,
                 float* __restrict__ enc, _Float16* __restrict__ enc16)
{
    const int b = blockIdx.x, d = threadIdx.x;
    if (d >= DIMV) {                        // keep f16 K-pad zero
        for (int t = 0; t < TART; ++t)
            enc16[((long long)(b * TART + t)) * KP + d] = (_Float16)0.f;
        return;
    }
    float w1[3][5], b1[3], w2[3];
    for (int k = 0; k < 3; ++k) {
        b1[k] = mr1b[k];
        w2[k] = mr2W[k];
        for (int r = 0; r < 5; ++r) w1[k][r] = mr1W[k * 5 + r];
    }
    const float b2 = mr2b[0];
    float c = 0.f;
    for (int t = 0; t < TART; ++t) {
        const long long idx = ((long long)(b * TART + t)) * DIMV + d;
        if (t + 8 < TART) {                 // stream lookahead
            const long long pidx = idx + 8LL * DIMV;
            __builtin_prefetch(z + pidx, 0, 3);
            __builtin_prefetch(o + pidx, 0, 3);
            __builtin_prefetch(h0 + pidx, 0, 3);
        }
        const float y0 = h0[((long long)(b * 2000 + t       )) * DIMV + d];
        const float y1 = h1[((long long)(b * 1000 + (t >> 1))) * DIMV + d] * 0.5f;
        const float y2 = h2[((long long)(b * 500  + (t >> 2))) * DIMV + d] * 0.25f;
        const float y3 = h3[((long long)(b * 200  + (t / 10))) * DIMV + d] * 0.1f;
        const float y4 = h4[((long long)(b * 80   + (t / 25))) * DIMV + d] * 0.04f;
        float g = b2;
        for (int k = 0; k < 3; ++k) {
            float hk = w1[k][0]*y0 + w1[k][1]*y1 + w1[k][2]*y2
                     + w1[k][3]*y3 + w1[k][4]*y4 + b1[k];
            hk = hk > 0.f ? hk : 0.f;
            g += w2[k] * hk;
        }
        g = g > 0.f ? g : 0.f;
        c = g * c + (1.f - g) * z[idx];
        const float e = o[idx] * c;
        enc[idx] = e;
        enc16[((long long)(b * TART + t)) * KP + d] = (_Float16)e;
    }
}

// --- pack f32 (nb,Tin,300) -> f16 (nb,Tpad,320), zero padded ---------------
__global__ __launch_bounds__(320)
void k_pack_rows(const float* __restrict__ in, _Float16* __restrict__ out,
                 int Tin, int Tpad)
{
    const int row = blockIdx.x;            // b*Tpad + tr
    const int c   = threadIdx.x;
    const int b = row / Tpad, tr = row % Tpad;
    const float v = (tr < Tin && c < DIMV)
                  ? in[((long long)(b * Tin + tr)) * DIMV + c] : 0.f;
    out[(long long)row * KP + c] = (_Float16)v;
}

// --- pack transposed values f32 (nb,Tw,300) -> f16 (nb,320,32) -------------
__global__ __launch_bounds__(32)
void k_pack_tr(const float* __restrict__ in, _Float16* __restrict__ out, int Tw)
{
    const int row = blockIdx.x;            // b*320 + d
    const int w   = threadIdx.x;           // 0..31
    const int b = row / 320, d = row % 320;
    const float v = (d < DIMV && w < Tw)
                  ? in[((long long)(b * Tw + w)) * DIMV + d] : 0.f;
    out[(long long)row * 32 + w] = (_Float16)v;
}

// --- row softmax over n valid of 32 cols; emits f16 probs (pad zero) -------
__global__ __launch_bounds__(256)
void k_softmax(const float* __restrict__ s, _Float16* __restrict__ p,
               int rows, int n)
{
    const int lane = threadIdx.x & 31;
    const int row  = blockIdx.x * 8 + (threadIdx.x >> 5);
    if (row >= rows) return;
    const float v = (lane < n) ? s[(long long)row * 32 + lane] : -1e30f;
    float m = v;
    for (int off = 16; off > 0; off >>= 1) m = fmaxf(m, __shfl_xor(m, off, 32));
    const float e = (lane < n) ? __expf(v - m) : 0.f;
    float sum = e;
    for (int off = 16; off > 0; off >>= 1) sum += __shfl_xor(sum, off, 32);
    p[(long long)row * 32 + lane] = (_Float16)(e / sum);
}

// --- mean over T of (B,T,300) into ans[:, colOff..colOff+300) --------------
__global__ __launch_bounds__(320)
void k_mean(const float* __restrict__ in, float* __restrict__ ansOpt, int colOff)
{
    const int b = blockIdx.x, d = threadIdx.x;
    if (d >= DIMV) return;
    float s = 0.f;
    const float* p = in + ((long long)b * TART) * DIMV + d;
    for (int t = 0; t < TART; ++t) s += p[(long long)t * DIMV];
    ansOpt[(long long)b * 600 + colOff + d] = s * (1.f / TART);
}

// --- answer MLP: relu(W1(75x600) vec + b1) -> W2(1x75) -> out[b*4+opt] -----
__global__ __launch_bounds__(128)
void k_answer(const float* __restrict__ ans, const float* __restrict__ W1,
              const float* __restrict__ b1, const float* __restrict__ W2,
              const float* __restrict__ b2, float* __restrict__ out)
{
    __shared__ float hsm[80];
    const int id  = blockIdx.x;            // opt*16 + b
    const int opt = id >> 4, b = id & 15;
    const float* vec = ans + (long long)id * 600;
    const int i = threadIdx.x;
    if (i < 75) {
        float s = b1[i];
        const float* wr = W1 + (long long)i * 600;
        for (int j = 0; j < 600; ++j) s += wr[j] * vec[j];
        hsm[i] = s > 0.f ? s : 0.f;
    }
    __syncthreads();
    if (i == 0) {
        float s = b2[0];
        for (int j = 0; j < 75; ++j) s += W2[j] * hsm[j];
        out[b * 4 + opt] = s;
    }
}

// ---------------------------------------------------------------------------
extern "C" void kernel_launch(void* const* d_in, const int* in_sizes, int n_in,
                              void* d_out, int out_size, void* d_ws, size_t ws_size,
                              hipStream_t stream)
{
    (void)in_sizes; (void)n_in; (void)out_size; (void)ws_size;

    const int* opt_in[4] = {(const int*)d_in[0], (const int*)d_in[1],
                            (const int*)d_in[2], (const int*)d_in[3]};
    const int*   q_in   = (const int*)d_in[4];
    const int*   art_in = (const int*)d_in[5];
    const float* emb    = (const float*)d_in[6];
    const float* ce_W   = (const float*)d_in[7];
    const float* ce_b   = (const float*)d_in[8];
    const float* mr1W   = (const float*)d_in[9];
    const float* mr1b   = (const float*)d_in[10];
    const float* mr2W   = (const float*)d_in[11];
    const float* mr2b   = (const float*)d_in[12];
    const float* Wz = (const float*)d_in[13]; const float* bz = (const float*)d_in[14];
    const float* Wo = (const float*)d_in[15]; const float* bo = (const float*)d_in[16];
    const float* f1W = (const float*)d_in[17]; const float* f1b = (const float*)d_in[18];
    const float* f2W = (const float*)d_in[19]; const float* f2b = (const float*)d_in[20];
    const float* f3W = (const float*)d_in[21]; const float* f3b = (const float*)d_in[22];
    const float* as1W = (const float*)d_in[23]; const float* as1b = (const float*)d_in[24];
    const float* as2W = (const float*)d_in[25]; const float* as2b = (const float*)d_in[26];
    float* out = (float*)d_out;

    // ---- workspace bump allocator (256B aligned) ----
    char* base = (char*)d_ws; size_t off = 0;
    auto alloc = [&](size_t bytes) -> void* {
        void* p = base + off;
        off = (off + bytes + 255) & ~(size_t)255;
        return p;
    };

    const int NA = BB * TART;                                  // 32000 rows
    static const int G[5]  = {2000, 1000, 500, 200, 80};
    static const int RR[5] = {1, 2, 4, 10, 25};

    // f32 buffers
    float* art32 = (float*)alloc((size_t)NA * 300 * 4);
    float* q32   = (float*)alloc((size_t)480 * 300 * 4);
    float* opt32 = (float*)alloc((size_t)1024 * 300 * 4);
    float* z32   = (float*)alloc((size_t)NA * 300 * 4);        // reused as aoqo
    float* o32   = (float*)alloc((size_t)NA * 300 * 4);        // reused as ooa
    float* hce[5];
    for (int i = 0; i < 5; ++i) hce[i] = (float*)alloc((size_t)BB * G[i] * 300 * 4);
    float* enc32 = (float*)alloc((size_t)NA * 300 * 4);
    float* aoq32 = (float*)alloc((size_t)NA * 300 * 4);
    float* sc32  = (float*)alloc((size_t)NA * 32 * 4);
    float* k1_32 = (float*)alloc((size_t)480 * 300 * 4);
    float* k2_32 = (float*)alloc((size_t)256 * 300 * 4);
    float* k3_32 = (float*)alloc((size_t)256 * 300 * 4);
    float* ans   = (float*)alloc((size_t)4 * BB * 600 * 4);

    // f16 buffers (all padded, pads written zero by producers)
    _Float16* art16 = (_Float16*)alloc((size_t)NA * KP * 2);   // reused as aoq16
    _Float16* q16   = (_Float16*)alloc((size_t)480 * KP * 2);
    _Float16* opt16 = (_Float16*)alloc((size_t)1024 * KP * 2);
    _Float16* xs16  = (_Float16*)alloc((size_t)NA * KP * 2);   // reused per range
    _Float16* w16[10];
    for (int i = 0; i < 10; ++i) w16[i] = (_Float16*)alloc((size_t)320 * KP * 2);
    _Float16* enc16 = (_Float16*)alloc((size_t)NA * KP * 2);
    _Float16* k1_16 = (_Float16*)alloc((size_t)(BB * 32) * KP * 2);
    _Float16* k2_16 = (_Float16*)alloc((size_t)(BB * 32) * KP * 2);
    _Float16* k3_16 = (_Float16*)alloc((size_t)(BB * 32) * KP * 2);
    _Float16* vtq   = (_Float16*)alloc((size_t)BB * 320 * 32 * 2);
    _Float16* vto   = (_Float16*)alloc((size_t)BB * 320 * 32 * 2);
    _Float16* pr16  = (_Float16*)alloc((size_t)NA * 32 * 2);

    // NT=4 for wide-N GEMMs, NT=2 for attention scores; Kp in {320, 32}.
    auto gemm = [&](const _Float16* A, long long as_, const _Float16* W, long long wsb,
                    const float* bias, float* C, long long cs,
                    int M, int N, int Kp, int ldC, int act, int nb, int NT) {
        const int mblk   = ((M / 16) + 3) / 4;      // 4 M-tiles (waves) / block
        const int ntiles = (N + 15) / 16;
        if (NT == 4 && Kp == KP) {
            dim3 grid(mblk, (ntiles + 3) / 4, nb);
            k_gemm_wmma<4, KP><<<grid, dim3(128), 0, stream>>>(A, as_, W, wsb, bias,
                                                               C, cs, M, N, ldC, act);
        } else if (NT == 4) {                        // Kp == 32 (probs @ values)
            dim3 grid(mblk, (ntiles + 3) / 4, nb);
            k_gemm_wmma<4, 32><<<grid, dim3(128), 0, stream>>>(A, as_, W, wsb, bias,
                                                               C, cs, M, N, ldC, act);
        } else {                                     // NT == 2, Kp == 320 (scores)
            dim3 grid(mblk, (ntiles + 1) / 2, nb);
            k_gemm_wmma<2, KP><<<grid, dim3(128), 0, stream>>>(A, as_, W, wsb, bias,
                                                               C, cs, M, N, ldC, act);
        }
    };

    // 1) embedding gathers
    k_gather<<<NA,  320, 0, stream>>>(art_in, emb, art32, art16);
    k_gather<<<480, 320, 0, stream>>>(q_in,   emb, q32,   q16);
    for (int i = 0; i < 4; ++i)
        k_gather<<<256, 320, 0, stream>>>(opt_in[i], emb,
                                          opt32 + (size_t)i * 256 * 300,
                                          opt16 + (size_t)i * 256 * KP);

    // 2) cast weights: [0..4]=ce_W, 5=Wz, 6=Wo, 7=f1, 8=f2, 9=f3
    for (int i = 0; i < 5; ++i)
        k_cast_w<<<320, 320, 0, stream>>>(ce_W + (size_t)i * 300 * 300, w16[i]);
    k_cast_w<<<320, 320, 0, stream>>>(Wz,  w16[5]);
    k_cast_w<<<320, 320, 0, stream>>>(Wo,  w16[6]);
    k_cast_w<<<320, 320, 0, stream>>>(f1W, w16[7]);
    k_cast_w<<<320, 320, 0, stream>>>(f2W, w16[8]);
    k_cast_w<<<320, 320, 0, stream>>>(f3W, w16[9]);

    // 3) ContractExpand: group-sum + relu GEMM per range
    for (int i = 0; i < 5; ++i) {
        const int rows = BB * G[i];
        k_group_sum<<<rows, 320, 0, stream>>>(art32, xs16, RR[i], G[i]);
        gemm(xs16, 0, w16[i], 0, ce_b + i * 300, hce[i], 0,
             rows, 300, KP, 300, /*relu*/1, 1, 4);
    }

    // 4) z = tanh(art Wz^T + bz), o = tanh(art Wo^T + bo)
    gemm(art16, 0, w16[5], 0, bz, z32, 0, NA, 300, KP, 300, /*tanh*/2, 1, 4);
    gemm(art16, 0, w16[6], 0, bo, o32, 0, NA, 300, KP, 300, /*tanh*/2, 1, 4);

    // 5) gate MLP + MRU scan -> art_enc (f32 + padded f16)
    k_gate_scan<<<BB, 320, 0, stream>>>(hce[0], hce[1], hce[2], hce[3], hce[4],
                                        z32, o32, mr1W, mr1b, mr2W, mr2b,
                                        enc32, enc16);

    // 6) stage-1 bi-attention: article on question -> aoq
    gemm(q16, 0, w16[7], 0, f1b, k1_32, 0, 480, 300, KP, 300, 0, 1, 4);
    k_pack_rows<<<BB * 32, 320, 0, stream>>>(k1_32, k1_16, TQ, 32);
    k_pack_tr<<<BB * 320, 32, 0, stream>>>(q32, vtq, TQ);
    gemm(enc16, (long long)TART * KP, k1_16, 32LL * KP, nullptr,
         sc32, (long long)TART * 32, TART, TQ, KP, 32, 0, BB, 2);
    k_softmax<<<NA / 8, 256, 0, stream>>>(sc32, pr16, NA, TQ);
    gemm(pr16, (long long)TART * 32, vtq, 320LL * 32, nullptr,
         aoq32, (long long)TART * 300, TART, 300, 32, 300, 0, BB, 4);
    _Float16* aoq16 = art16;                       // art16 no longer needed
    k_pack_rows<<<NA, 320, 0, stream>>>(aoq32, aoq16, NA, NA);

    // 7) per-option attentions + means
    for (int i = 0; i < 4; ++i) {
        const float*    ov32 = opt32 + (size_t)i * 256 * 300;
        const _Float16* ov16 = opt16 + (size_t)i * 256 * KP;
        gemm(ov16, 0, w16[8], 0, f2b, k2_32, 0, 256, 300, KP, 300, 0, 1, 4);
        gemm(ov16, 0, w16[9], 0, f3b, k3_32, 0, 256, 300, KP, 300, 0, 1, 4);
        k_pack_rows<<<BB * 32, 320, 0, stream>>>(k2_32, k2_16, TOPT, 32);
        k_pack_rows<<<BB * 32, 320, 0, stream>>>(k3_32, k3_16, TOPT, 32);
        k_pack_tr<<<BB * 320, 32, 0, stream>>>(ov32, vto, TOPT);

        // aoqo = attn(aoq, opt; f2)
        gemm(aoq16, (long long)TART * KP, k2_16, 32LL * KP, nullptr,
             sc32, (long long)TART * 32, TART, TOPT, KP, 32, 0, BB, 2);
        k_softmax<<<NA / 8, 256, 0, stream>>>(sc32, pr16, NA, TOPT);
        gemm(pr16, (long long)TART * 32, vto, 320LL * 32, nullptr,
             z32, (long long)TART * 300, TART, 300, 32, 300, 0, BB, 4);
        k_mean<<<BB, 320, 0, stream>>>(z32, ans + (size_t)i * BB * 600, 0);

        // ooa = attn(aoq, opt; f3)
        gemm(aoq16, (long long)TART * KP, k3_16, 32LL * KP, nullptr,
             sc32, (long long)TART * 32, TART, TOPT, KP, 32, 0, BB, 2);
        k_softmax<<<NA / 8, 256, 0, stream>>>(sc32, pr16, NA, TOPT);
        gemm(pr16, (long long)TART * 32, vto, 320LL * 32, nullptr,
             o32, (long long)TART * 300, TART, 300, 32, 300, 0, BB, 4);
        k_mean<<<BB, 320, 0, stream>>>(o32, ans + (size_t)i * BB * 600, 300);
    }

    // 8) answer selection MLP -> (B,4)
    k_answer<<<64, 128, 0, stream>>>(ans, as1W, as1b, as2W, as2b, out);
}
